// PWCRGBDNet_33844342293226
// MI455X (gfx1250) — matrix operand order repaired and checked by reading, not compiled, not run on hardware
//
#include <hip/hip_runtime.h>
#include <stdint.h>
#include <stddef.h>

typedef _Float16 half_t;
typedef __attribute__((ext_vector_type(16))) _Float16 v16h;
typedef __attribute__((ext_vector_type(8)))  float    v8f;

#define THREADS 256
static inline int cdiv(int a, int b) { return (a + b - 1) / b; }

// ---------------------------------------------------------------------------
// Weight packing: OIHW f32 -> f16 [Npad16][Kpad32] with K = (ky*3+kx)*Cin + ci
// (tap-major, channel-minor) to match NHWC activations.
// ---------------------------------------------------------------------------
__global__ void k_pack_w(const float* __restrict__ w, half_t* __restrict__ out,
                         int Cout, int Cin, int K, int Kpad, int total) {
  int i = blockIdx.x * blockDim.x + threadIdx.x;
  if (i >= total) return;
  int n = i / Kpad, k = i - n * Kpad;
  float v = 0.0f;
  if (n < Cout && k < K) {
    int t = k / Cin, ci = k - t * Cin;
    v = w[((size_t)n * Cin + ci) * 9 + t];
  }
  out[i] = (half_t)v;
}

// NCHW f32 -> NHWC f16 with channel placement
__global__ void k_cast_nchw_place_f16(const float* __restrict__ src, half_t* __restrict__ dst,
                                      int Bc, int C, int HW, int coff, int ctot) {
  int i = blockIdx.x * blockDim.x + threadIdx.x;
  if (i >= Bc * C * HW) return;
  int b = i / (C * HW); int r = i - b * C * HW;
  int c = r / HW;       int p = r - c * HW;
  dst[((size_t)b * HW + p) * ctot + coff + c] = (half_t)src[i];
}

// NHWC f32 (C=3, e.g. flow) -> NHWC f16 with channel placement
__global__ void k_cast_nhwc_place_f16(const float* __restrict__ src, half_t* __restrict__ dst,
                                      int n, int coff, int ctot) {
  int i = blockIdx.x * blockDim.x + threadIdx.x;
  if (i >= n) return;
  int q = i / 3, c = i - q * 3;
  dst[(size_t)q * ctot + coff + c] = (half_t)src[i];
}

// NHWC channel-slice copy
__global__ void k_copy_ch_f16(const half_t* __restrict__ src, half_t* __restrict__ dst,
                              int nPix, int C, int coffS, int ctotS, int coffD, int ctotD) {
  int i = blockIdx.x * blockDim.x + threadIdx.x;
  if (i >= nPix * C) return;
  int q = i / C, c = i - q * C;
  dst[(size_t)q * ctotD + coffD + c] = src[(size_t)q * ctotS + coffS + c];
}

__global__ void k_zero_f32(float* __restrict__ p, int n) {
  int i = blockIdx.x * blockDim.x + threadIdx.x;
  if (i < n) p[i] = 0.0f;
}

__global__ void k_add3_f32(const float* __restrict__ a, const float* __restrict__ b,
                           const float* __restrict__ c, float* __restrict__ o, int n) {
  int i = blockIdx.x * blockDim.x + threadIdx.x;
  if (i < n) o[i] = a[i] + b[i] + c[i];
}

// ---------------------------------------------------------------------------
// Align-corners bilinear resize, NHWC f32 -> NHWC f32, values * scale (flow).
// ---------------------------------------------------------------------------
__global__ void k_resize_f32(const float* __restrict__ src, float* __restrict__ dst,
                             int Bc, int C, int hs, int wsz, int hd, int wd, float scale) {
  int i = blockIdx.x * blockDim.x + threadIdx.x;
  if (i >= Bc * hd * wd * C) return;
  int c = i % C; int q = i / C;
  int ox = q % wd; int t = q / wd;
  int oy = t % hd; int b = t / hd;
  float fy = (hd > 1) ? oy * (float)(hs - 1) / (float)(hd - 1) : 0.0f;
  float fx = (wd > 1) ? ox * (float)(wsz - 1) / (float)(wd - 1) : 0.0f;
  int y0 = (int)fy, x0 = (int)fx;
  int y1 = y0 + 1 < hs ? y0 + 1 : hs - 1;
  int x1 = x0 + 1 < wsz ? x0 + 1 : wsz - 1;
  float wy = fy - (float)y0, wx = fx - (float)x0;
  const float* sp = src + (size_t)b * hs * wsz * C + c;
  float v = sp[((size_t)y0 * wsz + x0) * C] * (1.0f - wy) * (1.0f - wx)
          + sp[((size_t)y0 * wsz + x1) * C] * (1.0f - wy) * wx
          + sp[((size_t)y1 * wsz + x0) * C] * wy * (1.0f - wx)
          + sp[((size_t)y1 * wsz + x1) * C] * wy * wx;
  dst[i] = v * scale;
}

// Final upsample: NHWC f32 flow -> NCHW f32 output, values * scale.
__global__ void k_resize_out_f32(const float* __restrict__ src, float* __restrict__ dst,
                                 int Bc, int C, int hs, int wsz, int hd, int wd, float scale) {
  int i = blockIdx.x * blockDim.x + threadIdx.x;
  if (i >= Bc * C * hd * wd) return;
  int b = i / (C * hd * wd); int r = i - b * C * hd * wd;
  int c = r / (hd * wd);     int p = r - c * (hd * wd);
  int oy = p / wd, ox = p - oy * wd;
  float fy = (hd > 1) ? oy * (float)(hs - 1) / (float)(hd - 1) : 0.0f;
  float fx = (wd > 1) ? ox * (float)(wsz - 1) / (float)(wd - 1) : 0.0f;
  int y0 = (int)fy, x0 = (int)fx;
  int y1 = y0 + 1 < hs ? y0 + 1 : hs - 1;
  int x1 = x0 + 1 < wsz ? x0 + 1 : wsz - 1;
  float wy = fy - (float)y0, wx = fx - (float)x0;
  const float* sp = src + (size_t)b * hs * wsz * C + c;
  float v = sp[((size_t)y0 * wsz + x0) * C] * (1.0f - wy) * (1.0f - wx)
          + sp[((size_t)y0 * wsz + x1) * C] * (1.0f - wy) * wx
          + sp[((size_t)y1 * wsz + x0) * C] * wy * (1.0f - wx)
          + sp[((size_t)y1 * wsz + x1) * C] * wy * wx;
  dst[i] = v * scale;
}

// NCHW f32 single-channel (disparity) -> NHWC f16 placed
__global__ void k_resize_place_f16(const float* __restrict__ src, half_t* __restrict__ dst,
                                   int Bc, int hs, int wsz, int hd, int wd,
                                   int coff, int ctot) {
  int i = blockIdx.x * blockDim.x + threadIdx.x;
  if (i >= Bc * hd * wd) return;
  int b = i / (hd * wd); int p = i - b * (hd * wd);
  int oy = p / wd, ox = p - oy * wd;
  float fy = (hd > 1) ? oy * (float)(hs - 1) / (float)(hd - 1) : 0.0f;
  float fx = (wd > 1) ? ox * (float)(wsz - 1) / (float)(wd - 1) : 0.0f;
  int y0 = (int)fy, x0 = (int)fx;
  int y1 = y0 + 1 < hs ? y0 + 1 : hs - 1;
  int x1 = x0 + 1 < wsz ? x0 + 1 : wsz - 1;
  float wy = fy - (float)y0, wx = fx - (float)x0;
  const float* sp = src + (size_t)b * hs * wsz;
  float v = sp[(size_t)y0 * wsz + x0] * (1.0f - wy) * (1.0f - wx)
          + sp[(size_t)y0 * wsz + x1] * (1.0f - wy) * wx
          + sp[(size_t)y1 * wsz + x0] * wy * (1.0f - wx)
          + sp[(size_t)y1 * wsz + x1] * wy * wx;
  dst[((size_t)b * hd * wd + p) * ctot + coff] = (half_t)v;
}

// ---------------------------------------------------------------------------
// Bilinear warp (NHWC f16) by NHWC flow[:, :2]; fuses last-channel depth add.
// ---------------------------------------------------------------------------
__global__ void k_warp_f16(const half_t* __restrict__ x, const float* __restrict__ fl,
                           half_t* __restrict__ out, int Bc, int C, int H, int W,
                           float depthf) {
  int i = blockIdx.x * blockDim.x + threadIdx.x;
  if (i >= Bc * H * W * C) return;
  int c = i % C; int q = i / C;        // q = global pixel index (b*H+y)*W+x
  int xp = q % W; int t = q / W;
  int y = t % H; int b = t / H;
  float px = (float)xp + fl[(size_t)q * 3 + 0];
  float py = (float)y + fl[(size_t)q * 3 + 1];
  float x0f = floorf(px), y0f = floorf(py);
  float wxv = px - x0f, wyv = py - y0f;
  const half_t* xb = x + (size_t)b * H * W * C + c;
  float acc = 0.0f;
#define PWC_CORNER(YF, XF, WGT)                                              \
  { float yf = (YF), xf = (XF);                                              \
    if (yf >= 0.0f && yf <= (float)(H - 1) && xf >= 0.0f && xf <= (float)(W - 1)) { \
      int yi = (int)yf, xi = (int)xf;                                        \
      acc += (float)xb[((size_t)yi * W + xi) * C] * (WGT); } }
  PWC_CORNER(y0f,        x0f,        (1.0f - wyv) * (1.0f - wxv))
  PWC_CORNER(y0f,        x0f + 1.0f, (1.0f - wyv) * wxv)
  PWC_CORNER(y0f + 1.0f, x0f,        wyv * (1.0f - wxv))
  PWC_CORNER(y0f + 1.0f, x0f + 1.0f, wyv * wxv)
#undef PWC_CORNER
  if (c == C - 1) acc += fl[(size_t)q * 3 + 2] * depthf;
  out[i] = (half_t)acc;
}

// ---------------------------------------------------------------------------
// 81-tap cost volume (search 4), NHWC: channel reduction reads contiguous
// halfs. Mean over C, fused lrelu(0.01), placed into concat buffer channels.
// ---------------------------------------------------------------------------
__global__ void k_costvol_f16(const half_t* __restrict__ f1, int ctot1,
                              const half_t* __restrict__ f2, int C,
                              half_t* __restrict__ out, int coff, int ctotO,
                              int Bc, int H, int W) {
  const int D = 4, S = 81, R = 9;
  int i = blockIdx.x * blockDim.x + threadIdx.x;
  if (i >= Bc * H * W * S) return;
  int s = i % S; int q = i / S;        // q = (b*H+y)*W+x
  int xx = q % W; int t = q / W;
  int y = t % H; int b = t / H;
  int si = s / R;
  int dy = si - D, dx = (s - si * R) - D;
  int y2 = y + dy, x2 = xx + dx;
  float acc = 0.0f;
  if ((unsigned)y2 < (unsigned)H && (unsigned)x2 < (unsigned)W) {
    const half_t* a  = f1 + (size_t)q * ctot1;
    const half_t* bb = f2 + ((size_t)(b * H + y2) * W + x2) * C;
    for (int c = 0; c < C; ++c)
      acc += (float)a[c] * (float)bb[c];
    acc /= (float)C;
  }
  acc = acc >= 0.0f ? acc : 0.01f * acc;
  out[(size_t)q * ctotO + coff + s] = (half_t)acc;
}

// ---------------------------------------------------------------------------
// Implicit-GEMM 3x3 conv via V_WMMA_F32_16X16X32_F16 (wave32), NHWC.
//   M = B*Ho*Wo, N = Cout, K = 9*Cin with k = (ky*3+kx)*Cin + ci.
//   Block = 128 threads = 4 waves, one shared M-tile (A in LDS, double
//   buffered, pipelined); each wave owns a 16x32 output tile (2 WMMAs/chunk,
//   A fragment reused). Gather has a uniform single-tap fast path (whole
//   chunk inside one 3x3 tap: contiguous 64B row loads, row-uniform bounds).
// ---------------------------------------------------------------------------
struct ConvArgs {
  const half_t* X; const half_t* Wp; const float* bias;
  half_t* Y16; float* Y32;
  int cinOff, cinTot, Cin, Hin, Win;
  int Kpad, K, M, Cout;
  int coutOff, coutTot, Hout, Wout;
  int stride, pad, dil, actv;
};

__global__ __launch_bounds__(128) void k_conv_wmma(ConvArgs A) {
  __shared__ alignas(32) half_t aT[2][16 * 32];
  const int tid  = threadIdx.x;
  const int wave = tid >> 5;
  const int lane = tid & 31;
  const int m0 = blockIdx.x << 4;
  const int n0 = blockIdx.y * 128 + wave * 32;
  const bool nv0 = n0 < A.Cout;
  const bool nv1 = (n0 + 16) < A.Cout;
  const int HW = A.Hout * A.Wout;
  const int WC = A.Win * A.cinTot;

  // Per-thread gather metadata: this thread fills rows ml = wave+4j, col = lane.
  int oyS[4], oxS[4], bBase[4]; bool mv[4];
#pragma unroll
  for (int j = 0; j < 4; ++j) {
    int mg = m0 + wave + 4 * j;
    mv[j] = mg < A.M;
    int mgc = mv[j] ? mg : 0;
    int b = mgc / HW; int r2 = mgc - b * HW;
    int oy = r2 / A.Wout; int ox = r2 - oy * A.Wout;
    oyS[j] = oy * A.stride - A.pad;
    oxS[j] = ox * A.stride - A.pad;
    bBase[j] = b * A.Hin * WC + A.cinOff;
  }

  const int aRow = lane & 15;
  const int kb   = (lane >> 4) << 4;
  const half_t* bp0 = A.Wp + (size_t)(n0 + (lane & 15)) * A.Kpad + ((lane >> 4) << 4);
  const half_t* bp1 = bp0 + (size_t)16 * A.Kpad;

  v8f acc0 = {0.f, 0.f, 0.f, 0.f, 0.f, 0.f, 0.f, 0.f};
  v8f acc1 = {0.f, 0.f, 0.f, 0.f, 0.f, 0.f, 0.f, 0.f};

  auto gather = [&](int klo, int buf) {
    int t0  = klo / A.Cin;            // uniform (klo is uniform)
    int rem = klo - t0 * A.Cin;
    if (rem + 32 <= A.Cin && klo + 32 <= A.K) {
      // Fast path: entire chunk inside tap t0 -> contiguous 64B row loads,
      // bounds depend only on row j (uniform across the wave).
      int kyd = (t0 / 3) * A.dil, kxd = (t0 - (t0 / 3) * 3) * A.dil;
      int ci = rem + lane;
#pragma unroll
      for (int j = 0; j < 4; ++j) {
        half_t v = (half_t)0.0f;
        int iy = oyS[j] + kyd, ix = oxS[j] + kxd;
        if (mv[j] && (unsigned)iy < (unsigned)A.Hin && (unsigned)ix < (unsigned)A.Win)
          v = A.X[(size_t)(bBase[j] + iy * WC + ix * A.cinTot + ci)];
        aT[buf][(wave + 4 * j) * 32 + lane] = v;
      }
      return;
    }
    int k = klo + lane;
    if (k < A.K) {
      int t = k / A.Cin; int ci = k - t * A.Cin;
      int kyd = (t / 3) * A.dil, kxd = (t - (t / 3) * 3) * A.dil;
#pragma unroll
      for (int j = 0; j < 4; ++j) {
        half_t v = (half_t)0.0f;
        int iy = oyS[j] + kyd, ix = oxS[j] + kxd;
        if (mv[j] && (unsigned)iy < (unsigned)A.Hin && (unsigned)ix < (unsigned)A.Win)
          v = A.X[(size_t)(bBase[j] + iy * WC + ix * A.cinTot + ci)];
        aT[buf][(wave + 4 * j) * 32 + lane] = v;
      }
    } else {
#pragma unroll
      for (int j = 0; j < 4; ++j)
        aT[buf][(wave + 4 * j) * 32 + lane] = (half_t)0.0f;
    }
  };

  const int nch = (A.K + 31) >> 5;
  gather(0, 0);
  __syncthreads();
#pragma unroll 2
  for (int c = 0; c < nch; ++c) {
    if (c + 1 < nch) gather((c + 1) << 5, (c + 1) & 1);
    if (c + 2 < nch) {
      if (nv0) __builtin_prefetch(bp0 + ((c + 2) << 5), 0, 0);  // global_prefetch_b8
      if (nv1) __builtin_prefetch(bp1 + ((c + 2) << 5), 0, 0);
    }
    // A fragment, contiguous k order k = 16h + e (matches B loads)
    v16h af;
    const half_t* as = &aT[c & 1][aRow * 32 + kb];
#pragma unroll
    for (int e = 0; e < 16; ++e)
      af[e] = as[e];
    const int klo = c << 5;
    if (nv0) {
      v16h bf0 = *(const v16h*)(bp0 + klo);
      acc0 = __builtin_amdgcn_wmma_f32_16x16x32_f16(false, af, false, bf0,
                                                    (short)0, acc0, false, false);
    }
    if (nv1) {
      v16h bf1 = *(const v16h*)(bp1 + klo);
      acc1 = __builtin_amdgcn_wmma_f32_16x16x32_f16(false, af, false, bf1,
                                                    (short)0, acc1, false, false);
    }
    __syncthreads();
  }

  // Epilogue: C/D layout -> NHWC. mg IS the pixel index (M = B*HW), so the
  // store offset is mg*ctot + n with consecutive n across lanes (coalesced).
  const int mBase = m0 + ((lane >> 4) << 3);
  for (int h = 0; h < 2; ++h) {
    int nCol = n0 + h * 16 + (lane & 15);
    bool nv = h ? nv1 : nv0;
    if (!nv || nCol >= A.Cout) continue;
    v8f& acc = h ? acc1 : acc0;
    float bv = A.bias ? A.bias[nCol] : 0.0f;
#pragma unroll
    for (int i = 0; i < 8; ++i) {
      int mg = mBase + i;
      if (mg < A.M) {
        float v = acc[i] + bv;
        if (A.actv) v = v >= 0.0f ? v : 0.1f * v;
        size_t o = (size_t)mg * A.coutTot + A.coutOff + nCol;
        if (A.Y32) A.Y32[o] = v;
        if (A.Y16) A.Y16[o] = (half_t)v;
      }
    }
  }
}

// ---------------------------------------------------------------------------
// Host orchestration
// ---------------------------------------------------------------------------
struct Layer {
  const float* w; const float* bias;
  int cin, cout, K, Kpad, Npad;
  half_t* pk;
};

struct Bump { char* p; size_t left; };
static void* balloc(Bump& b, size_t bytes) {
  size_t a = (bytes + 255) & ~(size_t)255;
  void* r = (void*)b.p;
  b.p += a;
  b.left = (b.left > a) ? (b.left - a) : 0;
  return r;
}

static void launch_conv(hipStream_t st, const Layer& L,
                        const half_t* X, int cinOff, int cinTot, int Hin, int Win,
                        half_t* Y16, float* Y32, int coutOff, int coutTot,
                        int Hout, int Wout, int stride, int pad, int dil, int actv,
                        int Bn) {
  ConvArgs A;
  A.X = X; A.Wp = L.pk; A.bias = L.bias; A.Y16 = Y16; A.Y32 = Y32;
  A.cinOff = cinOff; A.cinTot = cinTot; A.Cin = L.cin; A.Hin = Hin; A.Win = Win;
  A.Kpad = L.Kpad; A.K = L.K; A.M = Bn * Hout * Wout; A.Cout = L.cout;
  A.coutOff = coutOff; A.coutTot = coutTot; A.Hout = Hout; A.Wout = Wout;
  A.stride = stride; A.pad = pad; A.dil = dil; A.actv = actv;
  dim3 g(cdiv(A.M, 16), cdiv(cdiv(L.cout, 16), 8));
  k_conv_wmma<<<g, 128, 0, st>>>(A);
}

extern "C" void kernel_launch(void* const* d_in, const int* in_sizes, int n_in,
                              void* d_out, int out_size, void* d_ws, size_t ws_size,
                              hipStream_t stream) {
  (void)in_sizes; (void)out_size;
  const int Bn = 4;
  static const int LV[7]  = {3, 16, 32, 64, 96, 128, 192};
  const int lch[5] = {192, 128, 96, 64, 32};
  const int lh[5]  = {8, 16, 32, 64, 128};
  if (n_in < 158 || !d_out || !d_ws) return;

  const float* x1r = (const float*)d_in[0];
  const float* x2r = (const float*)d_in[1];
  const float* d1r = (const float*)d_in[2];
  const float* d2r = (const float*)d_in[3];

  // Param leaves in JAX pytree order: dict keys sorted -> ctx, est, fpe.
  int idx = 4;
  Layer ctxL[5][7], estL[5][6], fpeL[6][2];
  for (int l = 0; l < 5; ++l) {
    int ch = lch[l];
    int dims[8] = {ch + 4, 128, 128, 128, 96, 64, 32, 3};
    for (int j = 0; j < 7; ++j) {
      ctxL[l][j].w    = (const float*)d_in[idx++];
      ctxL[l][j].bias = (const float*)d_in[idx++];
      ctxL[l][j].cin  = dims[j];
      ctxL[l][j].cout = dims[j + 1];
    }
  }
  for (int l = 0; l < 5; ++l) {
    int ch = lch[l];
    int dims[7] = {ch + 85, 128, 128, 96, 64, 32, 3};
    for (int j = 0; j < 6; ++j) {
      estL[l][j].w    = (const float*)d_in[idx++];
      estL[l][j].bias = (const float*)d_in[idx++];
      estL[l][j].cin  = dims[j];
      estL[l][j].cout = dims[j + 1];
    }
  }
  for (int s = 0; s < 6; ++s) {
    for (int j = 0; j < 2; ++j) {
      fpeL[s][j].w    = (const float*)d_in[idx++];
      fpeL[s][j].bias = (const float*)d_in[idx++];
      fpeL[s][j].cin  = (j == 0) ? LV[s] : LV[s + 1];
      fpeL[s][j].cout = LV[s + 1];
    }
  }

  Bump ws; ws.p = (char*)d_ws; ws.left = ws_size;

  // Pack all weights to f16 [Npad][Kpad], tap-major K
  auto packL = [&](Layer& L) {
    L.K = L.cin * 9;
    L.Kpad = (L.K + 31) & ~31;
    L.Npad = (L.cout + 15) & ~15;
    int total = L.Npad * L.Kpad;
    L.pk = (half_t*)balloc(ws, (size_t)total * sizeof(half_t));
    k_pack_w<<<cdiv(total, THREADS), THREADS, 0, stream>>>(L.w, L.pk, L.cout, L.cin, L.K, L.Kpad, total);
  };
  for (int l = 0; l < 5; ++l) for (int j = 0; j < 7; ++j) packL(ctxL[l][j]);
  for (int l = 0; l < 5; ++l) for (int j = 0; j < 6; ++j) packL(estL[l][j]);
  for (int s = 0; s < 6; ++s) for (int j = 0; j < 2; ++j) packL(fpeL[s][j]);

  // Activation buffers (all NHWC)
  half_t* xr16[2];
  xr16[0] = (half_t*)balloc(ws, (size_t)Bn * 3 * 512 * 512 * 2);
  xr16[1] = (half_t*)balloc(ws, (size_t)Bn * 3 * 512 * 512 * 2);
  half_t* st1[2];
  st1[0] = (half_t*)balloc(ws, (size_t)Bn * 16 * 256 * 256 * 2);
  st1[1] = (half_t*)balloc(ws, (size_t)Bn * 16 * 256 * 256 * 2);
  half_t* tmpA = (half_t*)balloc(ws, (size_t)Bn * 16 * 256 * 256 * 2);
  half_t* estin[5]; half_t* f2cB[5];
  for (int l = 0; l < 5; ++l) {
    estin[l] = (half_t*)balloc(ws, (size_t)Bn * (lch[l] + 85) * lh[l] * lh[l] * 2);
    f2cB[l]  = (half_t*)balloc(ws, (size_t)Bn * (lch[l] + 1)  * lh[l] * lh[l] * 2);
  }
  half_t* ctxin = (half_t*)balloc(ws, (size_t)Bn * 36 * 128 * 128 * 2);
  half_t* f2w   = (half_t*)balloc(ws, (size_t)Bn * 33 * 128 * 128 * 2);
  half_t* pingH = (half_t*)balloc(ws, (size_t)Bn * 128 * 128 * 128 * 2);
  half_t* pongH = (half_t*)balloc(ws, (size_t)Bn * 128 * 128 * 128 * 2);
  float* estHead = (float*)balloc(ws, (size_t)Bn * 3 * 128 * 128 * 4);
  float* ctxHead = (float*)balloc(ws, (size_t)Bn * 3 * 128 * 128 * 4);
  float* flowA   = (float*)balloc(ws, (size_t)Bn * 3 * 128 * 128 * 4);
  float* flowB   = (float*)balloc(ws, (size_t)Bn * 3 * 128 * 128 * 4);

  // Cast raw RGB (NCHW f32) to NHWC f16
  {
    int n = Bn * 3 * 512 * 512;
    k_cast_nchw_place_f16<<<cdiv(n, THREADS), THREADS, 0, stream>>>(x1r, xr16[0], Bn, 3, 512 * 512, 0, 3);
    k_cast_nchw_place_f16<<<cdiv(n, THREADS), THREADS, 0, stream>>>(x2r, xr16[1], Bn, 3, 512 * 512, 0, 3);
  }

  // Feature pyramid (both images); stage s conv2 writes directly into the
  // per-level NHWC concat buffers (feat channels [0, ch)).
  for (int img = 0; img < 2; ++img) {
    const half_t* cur = xr16[img];
    int cTot = 3, H = 512, W = 512;
    for (int s = 1; s <= 6; ++s) {
      int oh = 512 >> s;
      launch_conv(stream, fpeL[s - 1][0], cur, 0, cTot, H, W,
                  tmpA, nullptr, 0, LV[s], oh, oh, 2, 1, 1, 1, Bn);
      half_t* dst; int dTot;
      if (s == 1) { dst = st1[img]; dTot = 16; }
      else {
        int l = 6 - s;
        dTot = (img == 0) ? lch[l] + 85 : lch[l] + 1;
        dst  = (img == 0) ? estin[l] : f2cB[l];
      }
      launch_conv(stream, fpeL[s - 1][1], tmpA, 0, LV[s], oh, oh,
                  dst, nullptr, 0, dTot, oh, oh, 1, 1, 1, 1, Bn);
      cur = dst; cTot = dTot; H = oh; W = oh;
    }
  }

  // Coarse-to-fine estimation
  float* fA = flowA; float* fB = flowB;
  const int ctxDil[7] = {1, 2, 4, 8, 16, 1, 1};
  half_t* bufs[2] = {pingH, pongH};
  for (int l = 0; l < 5; ++l) {
    int ch = lch[l], h = lh[l], w = h, HW = h * w;
    int Cc = ch + 1, CtotE = ch + 85, CtotC = ch + 4;
    int n3 = Bn * 3 * HW;

    if (l == 0) {
      k_zero_f32<<<cdiv(n3, THREADS), THREADS, 0, stream>>>(fA, n3);
    } else {
      k_resize_f32<<<cdiv(n3, THREADS), THREADS, 0, stream>>>(fA, fB, Bn, 3, h / 2, w / 2, h, w, 2.0f);
      float* t = fA; fA = fB; fB = t;
    }

    // disparities into concat buffers (channel ch)
    k_resize_place_f16<<<cdiv(Bn * HW, THREADS), THREADS, 0, stream>>>(
        d1r, estin[l], Bn, 512, 512, h, w, ch, CtotE);
    k_resize_place_f16<<<cdiv(Bn * HW, THREADS), THREADS, 0, stream>>>(
        d2r, f2cB[l], Bn, 512, 512, h, w, ch, Cc);

    // warp f2c by flow (fused depth-channel add)
    k_warp_f16<<<cdiv(Bn * Cc * HW, THREADS), THREADS, 0, stream>>>(
        f2cB[l], fA, f2w, Bn, Cc, h, w, 1.0f);

    // cost volume -> channels [Cc, Cc+81) of estin, fused lrelu(0.01)
    k_costvol_f16<<<cdiv(Bn * 81 * HW, THREADS), THREADS, 0, stream>>>(
        estin[l], CtotE, f2w, Cc, estin[l], Cc, CtotE, Bn, h, w);

    // flow -> f16 into estin tail and ctxin tail; f1c slice -> ctxin head
    k_cast_nhwc_place_f16<<<cdiv(n3, THREADS), THREADS, 0, stream>>>(
        fA, estin[l], n3, Cc + 81, CtotE);
    k_copy_ch_f16<<<cdiv(Bn * Cc * HW, THREADS), THREADS, 0, stream>>>(
        estin[l], ctxin, Bn * HW, Cc, 0, CtotE, 0, CtotC);
    k_cast_nhwc_place_f16<<<cdiv(n3, THREADS), THREADS, 0, stream>>>(
        fA, ctxin, n3, Cc, CtotC);

    // est stack (6 convs, dil=1, lrelu except last; head -> f32)
    {
      const half_t* in = estin[l]; int cT = CtotE;
      for (int j = 0; j < 6; ++j) {
        const Layer& L = estL[l][j];
        if (j < 5) {
          half_t* o = bufs[j & 1];
          launch_conv(stream, L, in, 0, cT, h, w, o, nullptr, 0, L.cout, h, w, 1, 1, 1, 1, Bn);
          in = o; cT = L.cout;
        } else {
          launch_conv(stream, L, in, 0, cT, h, w, nullptr, estHead, 0, 3, h, w, 1, 1, 1, 0, Bn);
        }
      }
    }
    // ctx stack (7 convs, dilated, lrelu except last; head -> f32)
    {
      const half_t* in = ctxin; int cT = CtotC;
      for (int j = 0; j < 7; ++j) {
        const Layer& L = ctxL[l][j];
        int d = ctxDil[j];
        if (j < 6) {
          half_t* o = bufs[j & 1];
          launch_conv(stream, L, in, 0, cT, h, w, o, nullptr, 0, L.cout, h, w, 1, d, d, 1, Bn);
          in = o; cT = L.cout;
        } else {
          launch_conv(stream, L, in, 0, cT, h, w, nullptr, ctxHead, 0, 3, h, w, 1, d, d, 0, Bn);
        }
      }
    }

    // flow = (est + flow) + ctx
    k_add3_f32<<<cdiv(n3, THREADS), THREADS, 0, stream>>>(estHead, ctxHead, fA, fB, n3);
    { float* t = fA; fA = fB; fB = t; }
  }

  // Final upsample x4 (128 -> 512), values scaled by 4, NHWC -> NCHW output
  {
    int n = Bn * 3 * 512 * 512;
    k_resize_out_f32<<<cdiv(n, THREADS), THREADS, 0, stream>>>(
        fA, (float*)d_out, Bn, 3, 128, 128, 512, 512, 4.0f);
  }
}
